// RelPosImTransformerEncoderLayer_7413113553635
// MI455X (gfx1250) — matrix operand order repaired and checked by reading, not compiled, not run on hardware
//
#include <hip/hip_runtime.h>
#include <hip/hip_bf16.h>

#define B_    4
#define LQ_   640
#define LIM_  1296
#define LK_   1936
#define LKP_  1952
#define DM_   768
#define NH_   12
#define DK_   64
#define DFF_  3072
#define MAXD_ 100

typedef __attribute__((ext_vector_type(16))) _Float16 v16h;
typedef __attribute__((ext_vector_type(8)))  _Float16 v8h;
typedef __attribute__((ext_vector_type(8)))  float    v8f;

#if defined(__has_builtin)
#  if __has_builtin(__builtin_amdgcn_global_load_async_to_lds_b32) && \
      __has_builtin(__builtin_amdgcn_s_wait_asynccnt)
#    define USE_ASYNC_LDS 1
#  endif
#endif
#ifndef USE_ASYNC_LDS
#  define USE_ASYNC_LDS 0
#endif

#if USE_ASYNC_LDS
typedef __attribute__((address_space(1))) int as1_int;
typedef __attribute__((address_space(3))) int as3_int;
#endif

static __device__ __forceinline__ int imin(int a, int b) { return a < b ? a : b; }
static __device__ __forceinline__ int iclip(int v, int lo, int hi) {
  return v < lo ? lo : (v > hi ? hi : v);
}

static __device__ __forceinline__ v8f wmma32(v16h a, v16h b, v8f c) {
  return __builtin_amdgcn_wmma_f32_16x16x32_f16(false, a, false, b, (short)0, c,
                                                false, false);
}

// A fragment (16x32 f16): lane holds row m = lane&15.
// element i<8  -> k = lh*8 + i        (chunk at k0 + lh*8)
// element i>=8 -> k = 16 + lh*8 + i-8 (chunk at k0 + 16 + lh*8)
static __device__ __forceinline__ v16h load_a32(const _Float16* __restrict__ A, int lda,
                                                int row, int maxrow, int k0, int lane) {
  const int m = lane & 15, lh = lane >> 4;
  int r = row + m;
  if (r > maxrow) r = maxrow;
  const _Float16* p = A + (size_t)r * lda + k0 + lh * 8;
  v8h lo = *(const v8h*)(p);
  v8h hi = *(const v8h*)(p + 16);
  v16h out;
#pragma unroll
  for (int i = 0; i < 8; ++i) { out[i] = lo[i]; out[i + 8] = hi[i]; }
  return out;
}

// B fragment (32x16 f16) from Bt stored [N][K] row-major.
// lane holds column n = lane&15; element i<8 -> k = lh*8 + i, i>=8 -> k = 16 + lh*8 + i-8
static __device__ __forceinline__ v16h load_b32(const _Float16* __restrict__ Bt, int ldb,
                                                int col, int k0, int lane) {
  const int n = lane & 15, lh = lane >> 4;
  const _Float16* p = Bt + (size_t)(col + n) * ldb + k0 + lh * 8;
  v8h lo = *(const v8h*)(p);
  v8h hi = *(const v8h*)(p + 16);
  v16h out;
#pragma unroll
  for (int i = 0; i < 8; ++i) { out[i] = lo[i]; out[i + 8] = hi[i]; }
  return out;
}

// ---------------------------------------------------------------------------
// Generic WMMA GEMM: C(MxN) = A(MxK, f16) * Bt(NxK, f16)^T + bias
// 256 threads = 8 waves; block tile 128x128; wave tile 32x64.
// MODE 0: f16 out, identity rows (RELU optional)
// MODE 1: f16 out, K-padded rows (b = m/LK_, row -> b*LKP_ + r)
// MODE 2: f16 out, V-transpose layout (b, h, dk, key) with LKP_ stride
// MODE 3: f32 out + residual
// ---------------------------------------------------------------------------
template <int MODE, int RELU>
__global__ void __launch_bounds__(256)
gemm_wmma(const _Float16* __restrict__ A, const _Float16* __restrict__ Bt,
          const float* __restrict__ bias, const float* __restrict__ resid,
          _Float16* __restrict__ outH, float* __restrict__ outF,
          int M, int N, int K) {
  const int lane = threadIdx.x & 31;
  const int w    = threadIdx.x >> 5;
  const int rowBase = blockIdx.y * 128 + (w >> 1) * 32;
  const int colBase = blockIdx.x * 128 + (w & 1) * 64;

  v8f acc[2][4];
#pragma unroll
  for (int r = 0; r < 2; ++r)
#pragma unroll
    for (int c = 0; c < 4; ++c)
#pragma unroll
      for (int j = 0; j < 8; ++j) acc[r][c][j] = 0.f;

  for (int k0 = 0; k0 < K; k0 += 32) {
    __builtin_prefetch(A + (size_t)(rowBase + (lane & 15)) * K + k0 + 128, 0, 1);
    v16h a0 = load_a32(A, K, rowBase,      M - 1, k0, lane);
    v16h a1 = load_a32(A, K, rowBase + 16, M - 1, k0, lane);
    v16h b0 = load_b32(Bt, K, colBase,      k0, lane);
    v16h b1 = load_b32(Bt, K, colBase + 16, k0, lane);
    v16h b2 = load_b32(Bt, K, colBase + 32, k0, lane);
    v16h b3 = load_b32(Bt, K, colBase + 48, k0, lane);
    acc[0][0] = wmma32(a0, b0, acc[0][0]);
    acc[0][1] = wmma32(a0, b1, acc[0][1]);
    acc[0][2] = wmma32(a0, b2, acc[0][2]);
    acc[0][3] = wmma32(a0, b3, acc[0][3]);
    acc[1][0] = wmma32(a1, b0, acc[1][0]);
    acc[1][1] = wmma32(a1, b1, acc[1][1]);
    acc[1][2] = wmma32(a1, b2, acc[1][2]);
    acc[1][3] = wmma32(a1, b3, acc[1][3]);
  }

  const int lh = lane >> 4;
  const int nn = lane & 15;
#pragma unroll
  for (int r = 0; r < 2; ++r) {
#pragma unroll
    for (int c = 0; c < 4; ++c) {
      const int n_g = colBase + c * 16 + nn;
      const float bn = bias[n_g];
#pragma unroll
      for (int j = 0; j < 8; ++j) {
        const int m_g = rowBase + r * 16 + j + 8 * lh;
        if (m_g >= M) continue;
        float v = acc[r][c][j] + bn;
        if (MODE == 0) {
          if (RELU) v = fmaxf(v, 0.f);
          outH[(size_t)m_g * N + n_g] = (_Float16)v;
        } else if (MODE == 1) {
          const int bb = m_g / LK_;            // compile-time divisor
          const int rr = m_g - bb * LK_;
          outH[(size_t)(bb * LKP_ + rr) * N + n_g] = (_Float16)v;
        } else if (MODE == 2) {
          const int bb  = m_g / LK_;
          const int key = m_g - bb * LK_;
          const int hh  = n_g >> 6;
          const int dk  = n_g & 63;
          outH[(((size_t)(bb * NH_ + hh)) * DK_ + dk) * (size_t)LKP_ + key] =
              (_Float16)v;
        } else {
          const size_t idx = (size_t)m_g * N + n_g;
          const float rv = resid ? resid[idx] : 0.f;
          outF[idx] = v + rv;
        }
      }
    }
  }
}

// ---------------------------------------------------------------------------
// Flash-style attention with rel-pos bias. One wave per (b, h, 16-query tile);
// all 8 waves of a block share one (b, h) (40 q-tiles per head, 8 | 40).
// Row sums are accumulated with an extra ones-column WMMA (cfrag[4]).
// ---------------------------------------------------------------------------
__global__ void __launch_bounds__(256)
attn_wmma(const _Float16* __restrict__ Qh, const _Float16* __restrict__ Kh,
          const _Float16* __restrict__ VT,
          const int* __restrict__ qx, const int* __restrict__ qy,
          const int* __restrict__ fx, const int* __restrict__ fy,
          const float* __restrict__ bx, const float* __restrict__ by,
          _Float16* __restrict__ ctx) {
  __shared__ _Float16 plds[8][16 * 32];
  __shared__ float s_bias[2 * (2 * MAXD_ + 1)];  // [0..200]=bias_x(:,h), [201..401]=bias_y(:,h)
  const int lane = threadIdx.x & 31;
  const int w    = threadIdx.x >> 5;
  const int lh   = lane >> 4;
  const int nn   = lane & 15;
  const int bh   = (int)(blockIdx.x * 8) / (LQ_ / 16);   // block-uniform
  const int qt   = (int)(blockIdx.x * 8 + w) % (LQ_ / 16);
  const int h    = bh % NH_;
  const int b    = bh / NH_;
  const int qbase = qt * 16;
  const int hoff  = h * DK_;

  // Stage the two per-head bias table columns into LDS.
  for (int i = threadIdx.x; i < 2 * (2 * MAXD_ + 1); i += 256) {
    const float* src = (i <= 2 * MAXD_) ? (bx + (size_t)i * NH_ + h)
                                        : (by + (size_t)(i - (2 * MAXD_ + 1)) * NH_ + h);
#if USE_ASYNC_LDS
    __builtin_amdgcn_global_load_async_to_lds_b32((as1_int*)src,
                                                  (as3_int*)&s_bias[i], 0, 0);
#else
    s_bias[i] = *src;
#endif
  }
#if USE_ASYNC_LDS
  __builtin_amdgcn_s_wait_asynccnt(0);
#endif
  __syncthreads();

  const _Float16* Qb = Qh + (size_t)b * LQ_ * DM_;
  const _Float16* Kb = Kh + (size_t)b * LKP_ * DM_;
  const _Float16* Vb = VT + ((size_t)b * NH_ + h) * DK_ * (size_t)LKP_;
  const int* qxb = qx + b * LQ_;
  const int* qyb = qy + b * LQ_;
  const int* fxb = fx + b * LK_;
  const int* fyb = fy + b * LK_;

  v16h qf0 = load_a32(Qb, DM_, qbase, LQ_ - 1, hoff,      lane);
  v16h qf1 = load_a32(Qb, DM_, qbase, LQ_ - 1, hoff + 32, lane);
#pragma unroll
  for (int i = 0; i < 16; ++i) {             // fold 1/sqrt(dk) = 0.125 into Q
    qf0[i] *= (_Float16)0.125f;
    qf1[i] *= (_Float16)0.125f;
  }
  v16h onesb;
#pragma unroll
  for (int i = 0; i < 16; ++i) onesb[i] = (_Float16)1.0f;

  int iqx[8], iqy[8];
#pragma unroll
  for (int j = 0; j < 8; ++j) {
    const int q = qbase + j + 8 * lh;
    iqx[j] = qxb[q];
    iqy[j] = qyb[q];
  }

  float mrow[8], alpha[8];
  v8f cfrag[5];  // [0..3] = ctx over d, [4] = row-sum accumulator (P @ ones)
#pragma unroll
  for (int j = 0; j < 8; ++j) mrow[j] = -1e30f;
#pragma unroll
  for (int t = 0; t < 5; ++t)
#pragma unroll
    for (int j = 0; j < 8; ++j) cfrag[t][j] = 0.f;

  for (int kk = 0; kk < LK_; kk += 32) {
    v8f s0, s1;
#pragma unroll
    for (int j = 0; j < 8; ++j) { s0[j] = 0.f; s1[j] = 0.f; }
    {
      v16h kb0 = load_b32(Kb, DM_, kk,      hoff,      lane);
      v16h kb1 = load_b32(Kb, DM_, kk,      hoff + 32, lane);
      v16h kb2 = load_b32(Kb, DM_, kk + 16, hoff,      lane);
      v16h kb3 = load_b32(Kb, DM_, kk + 16, hoff + 32, lane);
      s0 = wmma32(qf0, kb0, s0);
      s0 = wmma32(qf1, kb1, s0);
      s1 = wmma32(qf0, kb2, s1);
      s1 = wmma32(qf1, kb3, s1);
    }

    const int key0 = kk + nn;
    const int key1 = kk + 16 + nn;
    const bool val1 = key1 < LK_;
    const int kc0 = imin(key0, LK_ - 1);
    const int kc1 = imin(key1, LK_ - 1);
    const int kx0 = fxb[kc0], ky0 = fyb[kc0];
    const int kx1 = fxb[kc1], ky1 = fyb[kc1];

#pragma unroll
    for (int j = 0; j < 8; ++j) {
      float sv0 = s0[j] + s_bias[iclip(iqx[j] - kx0, -MAXD_, MAXD_) + MAXD_]
                        + s_bias[(2 * MAXD_ + 1) + iclip(iqy[j] - ky0, -MAXD_, MAXD_) + MAXD_];
      float sv1 = s1[j] + s_bias[iclip(iqx[j] - kx1, -MAXD_, MAXD_) + MAXD_]
                        + s_bias[(2 * MAXD_ + 1) + iclip(iqy[j] - ky1, -MAXD_, MAXD_) + MAXD_];
      if (!val1) sv1 = -1e30f;
      float mx = fmaxf(sv0, sv1);
#pragma unroll
      for (int d = 8; d >= 1; d >>= 1) mx = fmaxf(mx, __shfl_xor(mx, d, 32));
      const float mn = fmaxf(mrow[j], mx);
      alpha[j] = __expf(mrow[j] - mn);
      mrow[j]  = mn;
      plds[w][(j + 8 * lh) * 32 + nn]      = (_Float16)__expf(sv0 - mn);
      plds[w][(j + 8 * lh) * 32 + 16 + nn] =
          (_Float16)(val1 ? __expf(sv1 - mn) : 0.f);
    }

#pragma unroll
    for (int t = 0; t < 5; ++t)
#pragma unroll
      for (int j = 0; j < 8; ++j) cfrag[t][j] *= alpha[j];

    // Re-read P (16x32) from LDS in A-fragment layout (same-wave, no barrier).
    v8h plo = *(const v8h*)&plds[w][nn * 32 + lh * 8];
    v8h phi = *(const v8h*)&plds[w][nn * 32 + 16 + lh * 8];
    v16h pf;
#pragma unroll
    for (int i = 0; i < 8; ++i) { pf[i] = plo[i]; pf[i + 8] = phi[i]; }

    cfrag[4] = wmma32(pf, onesb, cfrag[4]);   // row sums
#pragma unroll
    for (int t = 0; t < 4; ++t) {
      v16h vb = load_b32(Vb, LKP_, t * 16, kk, lane);
      cfrag[t] = wmma32(pf, vb, cfrag[t]);
    }
  }

#pragma unroll
  for (int t = 0; t < 4; ++t) {
#pragma unroll
    for (int j = 0; j < 8; ++j) {
      const float val = cfrag[t][j] / cfrag[4][j];
      const int q = qbase + j + 8 * lh;
      ctx[((size_t)b * LQ_ + q) * DM_ + hoff + t * 16 + nn] = (_Float16)val;
    }
  }
}

// ---------------------------------------------------------------------------
// LayerNorm over 768 columns; one block per row; writes f32 (+ optional f16).
// ---------------------------------------------------------------------------
__global__ void __launch_bounds__(256)
layernorm_768(const float* __restrict__ x, const float* __restrict__ g,
              const float* __restrict__ be, float* __restrict__ outF,
              _Float16* __restrict__ outH) {
  __shared__ float red[256];
  const int t = threadIdx.x;
  const size_t row = blockIdx.x;
  const float* xr = x + row * DM_;
  const float v0 = xr[t], v1 = xr[t + 256], v2 = xr[t + 512];
  red[t] = v0 + v1 + v2;
  __syncthreads();
  for (int o = 128; o > 0; o >>= 1) {
    if (t < o) red[t] += red[t + o];
    __syncthreads();
  }
  const float mu = red[0] * (1.f / DM_);
  __syncthreads();
  const float d0 = v0 - mu, d1 = v1 - mu, d2 = v2 - mu;
  red[t] = d0 * d0 + d1 * d1 + d2 * d2;
  __syncthreads();
  for (int o = 128; o > 0; o >>= 1) {
    if (t < o) red[t] += red[t + o];
    __syncthreads();
  }
  const float rs = rsqrtf(red[0] * (1.f / DM_) + 1e-5f);
  const float o0 = d0 * rs * g[t] + be[t];
  const float o1 = d1 * rs * g[t + 256] + be[t + 256];
  const float o2 = d2 * rs * g[t + 512] + be[t + 512];
  outF[row * DM_ + t]       = o0;
  outF[row * DM_ + t + 256] = o1;
  outF[row * DM_ + t + 512] = o2;
  if (outH) {
    outH[row * DM_ + t]       = (_Float16)o0;
    outH[row * DM_ + t + 256] = (_Float16)o1;
    outH[row * DM_ + t + 512] = (_Float16)o2;
  }
}

// ------------------------------- prep kernels -------------------------------
__global__ void cvt_f16(const float* __restrict__ in, _Float16* __restrict__ out,
                        size_t n) {
  size_t i = (size_t)blockIdx.x * blockDim.x + threadIdx.x;
  const size_t stride = (size_t)gridDim.x * blockDim.x;
  for (; i < n; i += stride) out[i] = (_Float16)in[i];
}

__global__ void zero_f16(_Float16* __restrict__ p, size_t n) {
  size_t i = (size_t)blockIdx.x * blockDim.x + threadIdx.x;
  const size_t stride = (size_t)gridDim.x * blockDim.x;
  for (; i < n; i += stride) p[i] = (_Float16)0.f;
}

__global__ void build_full(const float* __restrict__ im, const float* __restrict__ dq,
                           _Float16* __restrict__ full, size_t n) {
  size_t i = (size_t)blockIdx.x * blockDim.x + threadIdx.x;
  const size_t stride = (size_t)gridDim.x * blockDim.x;
  for (; i < n; i += stride) {
    const size_t b = i / ((size_t)LK_ * DM_);
    const size_t rem = i - b * (size_t)LK_ * DM_;
    const size_t r = rem / DM_;
    const size_t c = rem - r * DM_;
    const float v = (r < LIM_) ? im[(b * LIM_ + r) * DM_ + c]
                               : dq[(b * LQ_ + (r - LIM_)) * DM_ + c];
    full[i] = (_Float16)v;
  }
}

__global__ void build_coords(const int* __restrict__ imx, const int* __restrict__ dqx,
                             const int* __restrict__ imy, const int* __restrict__ dqy,
                             int* __restrict__ fx, int* __restrict__ fy) {
  const int i = blockIdx.x * blockDim.x + threadIdx.x;
  if (i >= B_ * LK_) return;
  const int b = i / LK_, r = i - b * LK_;
  fx[i] = (r < LIM_) ? imx[b * LIM_ + r] : dqx[b * LQ_ + r - LIM_];
  fy[i] = (r < LIM_) ? imy[b * LIM_ + r] : dqy[b * LQ_ + r - LIM_];
}

// Transpose W (KxN f32) -> WT (NxK f16), 32x32 LDS tiles, block (32,8).
__global__ void wt_f16(const float* __restrict__ W, _Float16* __restrict__ WT,
                       int K, int N) {
  __shared__ float tile[32][33];
  const int kb = blockIdx.y * 32, nb = blockIdx.x * 32;
  for (int i = threadIdx.y; i < 32; i += 8)
    tile[i][threadIdx.x] = W[(size_t)(kb + i) * N + nb + threadIdx.x];
  __syncthreads();
  for (int i = threadIdx.y; i < 32; i += 8)
    WT[(size_t)(nb + i) * K + kb + threadIdx.x] = (_Float16)tile[threadIdx.x][i];
}

// ---------------------------------------------------------------------------
extern "C" void kernel_launch(void* const* d_in, const int* in_sizes, int n_in,
                              void* d_out, int out_size, void* d_ws, size_t ws_size,
                              hipStream_t stream) {
  (void)in_sizes; (void)n_in; (void)out_size; (void)ws_size;
  const float* docqa = (const float*)d_in[0];
  const float* im    = (const float*)d_in[1];
  const int* dqx = (const int*)d_in[2];
  const int* dqy = (const int*)d_in[3];
  const int* imx = (const int*)d_in[4];
  const int* imy = (const int*)d_in[5];
  const float* Wq = (const float*)d_in[6];   const float* bq = (const float*)d_in[7];
  const float* Wk = (const float*)d_in[8];   const float* bk = (const float*)d_in[9];
  const float* Wv = (const float*)d_in[10];  const float* bv = (const float*)d_in[11];
  const float* Wo = (const float*)d_in[12];  const float* bo = (const float*)d_in[13];
  const float* biasx = (const float*)d_in[14];
  const float* biasy = (const float*)d_in[15];
  const float* W1 = (const float*)d_in[16];  const float* b1 = (const float*)d_in[17];
  const float* W2 = (const float*)d_in[18];  const float* b2 = (const float*)d_in[19];
  const float* g1 = (const float*)d_in[20];  const float* be1 = (const float*)d_in[21];
  const float* g2 = (const float*)d_in[22];  const float* be2 = (const float*)d_in[23];
  float* out = (float*)d_out;

  char* p = (char*)d_ws;
  auto alloc = [&](size_t bytes) -> char* {
    char* r = p;
    p += (bytes + 255) & ~(size_t)255;
    return r;
  };
  const int MQ = B_ * LQ_;   // 2560
  const int MK = B_ * LK_;   // 7744
  _Float16* docqa_h = (_Float16*)alloc((size_t)MQ * DM_ * 2);
  _Float16* full_h  = (_Float16*)alloc((size_t)MK * DM_ * 2);
  int* fullx = (int*)alloc((size_t)B_ * LK_ * 4);
  int* fully = (int*)alloc((size_t)B_ * LK_ * 4);
  _Float16* WqT = (_Float16*)alloc((size_t)DM_ * DM_ * 2);
  _Float16* WkT = (_Float16*)alloc((size_t)DM_ * DM_ * 2);
  _Float16* WvT = (_Float16*)alloc((size_t)DM_ * DM_ * 2);
  _Float16* WoT = (_Float16*)alloc((size_t)DM_ * DM_ * 2);
  _Float16* W1T = (_Float16*)alloc((size_t)DM_ * DFF_ * 2);
  _Float16* W2T = (_Float16*)alloc((size_t)DM_ * DFF_ * 2);
  _Float16* Qhh = (_Float16*)alloc((size_t)MQ * DM_ * 2);
  _Float16* Khh = (_Float16*)alloc((size_t)B_ * LKP_ * DM_ * 2);
  _Float16* VTh = (_Float16*)alloc((size_t)B_ * NH_ * DK_ * LKP_ * 2);
  _Float16* ctxh = (_Float16*)alloc((size_t)MQ * DM_ * 2);
  float* tmpF = (float*)alloc((size_t)MQ * DM_ * 4);
  float* srcF = (float*)alloc((size_t)MQ * DM_ * 4);
  _Float16* srcH = (_Float16*)alloc((size_t)MQ * DM_ * 2);
  _Float16* h1 = (_Float16*)alloc((size_t)MQ * DFF_ * 2);
  float* tmp2 = (float*)alloc((size_t)MQ * DM_ * 4);

  // ---- prep ----
  cvt_f16<<<1024, 256, 0, stream>>>(docqa, docqa_h, (size_t)MQ * DM_);
  build_full<<<2048, 256, 0, stream>>>(im, docqa, full_h, (size_t)MK * DM_);
  build_coords<<<(B_ * LK_ + 255) / 256, 256, 0, stream>>>(imx, dqx, imy, dqy,
                                                           fullx, fully);
  dim3 tb(32, 8);
  wt_f16<<<dim3(DM_ / 32, DM_ / 32), tb, 0, stream>>>(Wq, WqT, DM_, DM_);
  wt_f16<<<dim3(DM_ / 32, DM_ / 32), tb, 0, stream>>>(Wk, WkT, DM_, DM_);
  wt_f16<<<dim3(DM_ / 32, DM_ / 32), tb, 0, stream>>>(Wv, WvT, DM_, DM_);
  wt_f16<<<dim3(DM_ / 32, DM_ / 32), tb, 0, stream>>>(Wo, WoT, DM_, DM_);
  wt_f16<<<dim3(DFF_ / 32, DM_ / 32), tb, 0, stream>>>(W1, W1T, DM_, DFF_);
  wt_f16<<<dim3(DM_ / 32, DFF_ / 32), tb, 0, stream>>>(W2, W2T, DFF_, DM_);
  zero_f16<<<2048, 256, 0, stream>>>(Khh, (size_t)B_ * LKP_ * DM_);
  zero_f16<<<2048, 256, 0, stream>>>(VTh, (size_t)B_ * NH_ * DK_ * LKP_);

  // ---- Q/K/V projections ----
  gemm_wmma<0, 0><<<dim3(DM_ / 128, (MQ + 127) / 128), 256, 0, stream>>>(
      docqa_h, WqT, bq, nullptr, Qhh, nullptr, MQ, DM_, DM_);
  gemm_wmma<1, 0><<<dim3(DM_ / 128, (MK + 127) / 128), 256, 0, stream>>>(
      full_h, WkT, bk, nullptr, Khh, nullptr, MK, DM_, DM_);
  gemm_wmma<2, 0><<<dim3(DM_ / 128, (MK + 127) / 128), 256, 0, stream>>>(
      full_h, WvT, bv, nullptr, VTh, nullptr, MK, DM_, DM_);

  // ---- attention ----
  attn_wmma<<<B_ * NH_ * (LQ_ / 16) / 8, 256, 0, stream>>>(
      Qhh, Khh, VTh, dqx, dqy, fullx, fully, biasx, biasy, ctxh);

  // ---- output projection + residual, LN1 ----
  gemm_wmma<3, 0><<<dim3(DM_ / 128, MQ / 128), 256, 0, stream>>>(
      ctxh, WoT, bo, docqa, nullptr, tmpF, MQ, DM_, DM_);
  layernorm_768<<<MQ, 256, 0, stream>>>(tmpF, g1, be1, srcF, srcH);

  // ---- FFN ----
  gemm_wmma<0, 1><<<dim3(DFF_ / 128, MQ / 128), 256, 0, stream>>>(
      srcH, W1T, b1, nullptr, h1, nullptr, MQ, DFF_, DM_);
  gemm_wmma<3, 0><<<dim3(DM_ / 128, MQ / 128), 256, 0, stream>>>(
      h1, W2T, b2, srcF, nullptr, tmp2, MQ, DM_, DFF_);
  layernorm_768<<<MQ, 256, 0, stream>>>(tmp2, g2, be2, out, nullptr);
}
